// RelativeAttention_35270271434811
// MI455X (gfx1250) — matrix operand order repaired and checked by reading, not compiled, not run on hardware
//
#include <hip/hip_runtime.h>
#include <hip/hip_bf16.h>

// ---------------------------------------------------------------------------
// RelativeAttention on MI455X (gfx1250, wave32, WMMA bf16 16x16x32,
// TDM tensor loads + async global->LDS staging)
// H=1024, NH=16, DH=64, RV=32, B=1, S=2048
// ---------------------------------------------------------------------------

typedef __attribute__((ext_vector_type(16))) __bf16 bf16x16;
typedef __attribute__((ext_vector_type(8)))  float  f32x8;
typedef __attribute__((ext_vector_type(4)))  float  f32x4;

// ---- bf16 <-> f32 helpers (raw ushort storage; RNE rounding) --------------
__device__ __forceinline__ unsigned short f2bfu(float f) {
    union { float f; unsigned u; } a; a.f = f;
    unsigned r = a.u + 0x7FFFu + ((a.u >> 16) & 1u);
    return (unsigned short)(r >> 16);
}
__device__ __forceinline__ float bfu2f(unsigned short s) {
    union { unsigned u; float f; } o; o.u = ((unsigned)s) << 16;
    return o.f;
}
__device__ __forceinline__ __bf16 ubf(unsigned short s) {
    union { unsigned short u; __bf16 b; } c; c.u = s;
    return c.b;
}
__device__ __forceinline__ void st_out(float* p, float v)          { *p = v; }
__device__ __forceinline__ void st_out(unsigned short* p, float v) { *p = f2bfu(v); }

// stage 4 elements (16B global granule -> 8B bf16 LDS granule)
__device__ __forceinline__ void stage4(unsigned short* dst, const float* src) {
    f32x4 v = *(const f32x4*)src;
    unsigned long long p =
          (unsigned long long)f2bfu(v[0])
        | ((unsigned long long)f2bfu(v[1]) << 16)
        | ((unsigned long long)f2bfu(v[2]) << 32)
        | ((unsigned long long)f2bfu(v[3]) << 48);
    *(unsigned long long*)dst = p;
}
__device__ __forceinline__ void stage4(unsigned short* dst, const unsigned short* src) {
    *(unsigned long long*)dst = *(const unsigned long long*)src;
}

// ---- CDNA5 async global -> LDS (ASYNCcnt), 128-bit per lane ----------------
__device__ __forceinline__ void async_ld16(void* lds, const void* gaddr) {
    unsigned l = (unsigned)(size_t)lds;                 // LDS addr = low 32 bits
    unsigned long long g = (unsigned long long)(size_t)gaddr;
    asm volatile("global_load_async_to_lds_b128 %0, %1, off"
                 :: "v"(l), "v"(g) : "memory");
}
__device__ __forceinline__ void wait_async0() {
    asm volatile("s_wait_asynccnt 0x0" ::: "memory");
}

// ---- CDNA5 TDM: DMA a 32x64 bf16 tile (row stride 1024 elems) into LDS -----
__device__ __forceinline__ void tdm_load_tile_32x64_bf16(unsigned ldsoff, const void* gbase) {
#if __has_builtin(__builtin_amdgcn_tensor_load_to_lds)
    typedef __attribute__((ext_vector_type(4))) unsigned u32x4;
    typedef __attribute__((ext_vector_type(8))) int      i32x8;
    typedef __attribute__((ext_vector_type(4))) int      i32x4;
    unsigned long long ga = (unsigned long long)(size_t)gbase;
    u32x4 g0;
    g0[0] = 1u;                                        // count=1, no gather
    g0[1] = ldsoff;                                    // lds_addr
    g0[2] = (unsigned)ga;                              // global_addr[31:0]
    g0[3] = ((unsigned)(ga >> 32) & 0x01FFFFFFu) | 0x80000000u;  // addr[56:32] | type=2
    i32x8 g1;
    g1[0] = 0x00010000;          // wg_mask=0, data_size=1 (2B)
    g1[1] = (int)(64u << 16);    // tensor_dim0 = 64
    g1[2] = (int)(4096u << 16);  // tensor_dim1 = 4096 (no OOB)
    g1[3] = (int)(64u << 16);    // tile_dim0 = 64
    g1[4] = 32;                  // tile_dim1 = 32, tile_dim2 = 0
    g1[5] = 1024;                // tensor_dim0_stride = 1024 elems
    g1[6] = 0;
    g1[7] = 0;
    i32x4 z4 = {0, 0, 0, 0};
    i32x8 z8 = {0, 0, 0, 0, 0, 0, 0, 0};
    // amdgpu-toolchain (clang-23) 6-arg form: (g0, g1, g2, g3, g4, cpol)
    __builtin_amdgcn_tensor_load_to_lds(g0, g1, z4, z4, z8, 0);
#else
    (void)ldsoff; (void)gbase;   // fallback handled at call site
#endif
}
__device__ __forceinline__ void wait_tensor0() {
#if __has_builtin(__builtin_amdgcn_s_wait_tensorcnt)
    __builtin_amdgcn_s_wait_tensorcnt((short)0);
#else
    asm volatile("s_wait_tensorcnt 0x0" ::: "memory");
#endif
}

// ---------------------------------------------------------------------------
// Tiled GEMM: C[M,N] = A[M,K] @ W[K,N] + bias[N]
// 128 threads = 4 waves; 64x64 tile; each wave: 16(M) x 64(N) = 4 accumulators
// ---------------------------------------------------------------------------
template <typename AT, typename OT>
__global__ void __launch_bounds__(128)
gemm64x64(const AT* __restrict__ A, const float* __restrict__ W,
          const float* __restrict__ bias, OT* __restrict__ C,
          int M, int N, int K)
{
    __shared__ unsigned short Ash[64 * 32];   // 64 rows x 32 k  (bf16)
    __shared__ unsigned short Bsh[32 * 64];   // 32 k   x 64 col (bf16)

    const int tid  = threadIdx.x;
    const int wave = tid >> 5, lane = tid & 31;
    const int m16  = lane & 15, g = lane >> 4;
    const int m0 = blockIdx.y * 64, n0 = blockIdx.x * 64;
    (void)M;

    f32x8 acc[4];
    #pragma unroll
    for (int t = 0; t < 4; ++t)
        #pragma unroll
        for (int r = 0; r < 8; ++r) acc[t][r] = 0.f;

    for (int k0 = 0; k0 < K; k0 += 32) {
        __syncthreads();
        #pragma unroll
        for (int i = 0; i < 4; ++i) {               // A: 64x32, 4-elem granules
            int e = (i * 128 + tid) * 4;
            stage4(&Ash[e], A + (size_t)(m0 + (e >> 5)) * K + k0 + (e & 31));
        }
        #pragma unroll
        for (int i = 0; i < 4; ++i) {               // B: 32x64
            int e = (i * 128 + tid) * 4;
            stage4(&Bsh[e], W + (size_t)(k0 + (e >> 6)) * N + n0 + (e & 63));
        }
        __syncthreads();

        // A fragment 16x32 bf16 (ISA 7.12.2): lanes 0-15 K=2v+8g, V4-7 K=16+...
        bf16x16 af;
        #pragma unroll
        for (int v = 0; v < 8; ++v) {
            int kl = (v < 4) ? (2 * v + 8 * g) : (16 + 2 * (v - 4) + 8 * g);
            af[2 * v]     = ubf(Ash[(16 * wave + m16) * 32 + kl]);
            af[2 * v + 1] = ubf(Ash[(16 * wave + m16) * 32 + kl + 1]);
        }
        #pragma unroll
        for (int t = 0; t < 4; ++t) {
            bf16x16 bfrag;                          // B fragment 32x16: K same as A fragment
            #pragma unroll
            for (int v = 0; v < 8; ++v) {
                int kl = (v < 4) ? (2 * v + 8 * g) : (16 + 2 * (v - 4) + 8 * g);
                bfrag[2 * v]     = ubf(Bsh[kl * 64 + 16 * t + m16]);
                bfrag[2 * v + 1] = ubf(Bsh[(kl + 1) * 64 + 16 * t + m16]);
            }
            acc[t] = __builtin_amdgcn_wmma_f32_16x16x32_bf16(
                false, af, false, bfrag, (short)0, acc[t], false, false);
        }
    }

    #pragma unroll
    for (int t = 0; t < 4; ++t)
        #pragma unroll
        for (int r = 0; r < 8; ++r) {
            int row = m0 + 16 * wave + r + 8 * g;
            int col = n0 + 16 * t + m16;
            st_out(C + (size_t)row * N + col, acc[t][r] + bias[col]);
        }
}

// ---------------------------------------------------------------------------
// all_rel[h][q][r] = dot(q[q,h,:], rel_emb[r,h,:]) + rel_bias[r,h]
// ---------------------------------------------------------------------------
__global__ void __launch_bounds__(256)
rel_kernel(const unsigned short* __restrict__ qm, const float* __restrict__ rel_emb,
           const float* __restrict__ rel_bias, float* __restrict__ all_rel)
{
    int idx = blockIdx.x * 256 + threadIdx.x;     // 2048*16*32 total
    int r = idx & 31;
    int h = (idx >> 5) & 15;
    int q = idx >> 9;
    float acc = rel_bias[r * 16 + h];
    const unsigned long long* qp = (const unsigned long long*)(qm + (size_t)q * 1024 + h * 64);
    const f32x4* ep = (const f32x4*)(rel_emb + ((size_t)r * 16 + h) * 64);
    #pragma unroll
    for (int d4 = 0; d4 < 16; ++d4) {
        unsigned long long qv = qp[d4];
        f32x4 ev = ep[d4];
        acc += bfu2f((unsigned short)(qv      )) * ev[0];
        acc += bfu2f((unsigned short)(qv >> 16)) * ev[1];
        acc += bfu2f((unsigned short)(qv >> 32)) * ev[2];
        acc += bfu2f((unsigned short)(qv >> 48)) * ev[3];
    }
    all_rel[((size_t)h * 2048 + q) * 32 + r] = acc;
}

// ---------------------------------------------------------------------------
// Flash attention with relative-score gather.
// Grid: (S/64, NH). Block: 128 = 4 waves; wave w owns q rows [q0+16w, +16).
// Per 32-key chunk: K/V tiles arrive via TDM (TENSORcnt), ids/mask tiles via
// async global->LDS (ASYNCcnt); rel slice + gather served from LDS.
// ---------------------------------------------------------------------------
__global__ void __launch_bounds__(128)
attn_kernel(const unsigned short* __restrict__ qm, const unsigned short* __restrict__ km,
            const unsigned short* __restrict__ vm, const float* __restrict__ all_rel,
            const int* __restrict__ ids, const int* __restrict__ msk,
            unsigned short* __restrict__ ctx)
{
    __shared__ unsigned short Ksh[32 * 64];       // 4 KB
    __shared__ unsigned short Vsh[32 * 64];       // 4 KB
    __shared__ float          relsh[64 * 32];     // 8 KB
    __shared__ unsigned short Psh[4][16 * 32];    // 4 KB wave-private P patches
    __shared__ int            idssh[64 * 32];     // 8 KB
    __shared__ int            msksh[64 * 32];     // 8 KB

    const int h  = blockIdx.y;
    const int q0 = blockIdx.x * 64;
    const int tid  = threadIdx.x;
    const int wave = tid >> 5, lane = tid & 31;
    const int n16  = lane & 15, g = lane >> 4;

    #pragma unroll
    for (int i = 0; i < 4; ++i) {                 // rel slice: 2048 f32, contig
        int e = (i * 128 + tid) * 4;
        *(f32x4*)&relsh[e] = *(const f32x4*)(all_rel + ((size_t)h * 2048 + q0) * 32 + e);
    }

    // Q fragments for this wave's 16 rows (registers, whole k loop)
    const int qrow = q0 + 16 * wave + n16;
    bf16x16 aq[2];
    #pragma unroll
    for (int hf = 0; hf < 2; ++hf)
        #pragma unroll
        for (int v = 0; v < 8; ++v) {
            int kl = (v < 4) ? (2 * v + 8 * g) : (16 + 2 * (v - 4) + 8 * g);
            int d  = 32 * hf + kl;
            aq[hf][2 * v]     = ubf(qm[(size_t)qrow * 1024 + h * 64 + d]);
            aq[hf][2 * v + 1] = ubf(qm[(size_t)qrow * 1024 + h * 64 + d + 1]);
        }

    float mrun[8], lrun[8];
    f32x8 o[4];
    #pragma unroll
    for (int r = 0; r < 8; ++r) { mrun[r] = -3.0e38f; lrun[r] = 0.f; }
    #pragma unroll
    for (int t = 0; t < 4; ++t)
        #pragma unroll
        for (int r = 0; r < 8; ++r) o[t][r] = 0.f;

    for (int kk = 0; kk < 2048; kk += 32) {
        __syncthreads();

        // ---- K/V tiles via Tensor Data Mover (one wave issues the DMA) ----
#if __has_builtin(__builtin_amdgcn_tensor_load_to_lds)
        if (wave == 0) {
            tdm_load_tile_32x64_bf16((unsigned)(size_t)(void*)Ksh,
                                     km + (size_t)kk * 1024 + h * 64);
            tdm_load_tile_32x64_bf16((unsigned)(size_t)(void*)Vsh,
                                     vm + (size_t)kk * 1024 + h * 64);
        }
#else
        #pragma unroll
        for (int i = 0; i < 2; ++i) {             // per-lane async fallback
            int slot = i * 128 + tid;             // 256 x 16B granules
            int r = slot >> 3, c = (slot & 7) * 8;
            async_ld16(&Ksh[r * 64 + c], km + (size_t)(kk + r) * 1024 + h * 64 + c);
            async_ld16(&Vsh[r * 64 + c], vm + (size_t)(kk + r) * 1024 + h * 64 + c);
        }
#endif
        // ---- ids/mask tiles via async global->LDS (row-contiguous 128B) ----
        #pragma unroll
        for (int i = 0; i < 4; ++i) {
            int slot = i * 128 + tid;             // 512 x 16B granules each
            int r = slot >> 3, c = (slot & 7) * 4;
            async_ld16(&idssh[r * 32 + c], ids + (size_t)(q0 + r) * 2048 + kk + c);
            async_ld16(&msksh[r * 32 + c], msk + (size_t)(q0 + r) * 2048 + kk + c);
        }
        if (kk + 32 < 2048)
            __builtin_prefetch(ids + (size_t)qrow * 2048 + kk + 32, 0, 0);

        wait_async0();
        if (wave == 0) wait_tensor0();
        __syncthreads();

        // ---- S = Q @ K^T : two 16-key tiles, d split in 2 halves ----
        f32x8 s[2];
        #pragma unroll
        for (int t = 0; t < 2; ++t)
            #pragma unroll
            for (int r = 0; r < 8; ++r) s[t][r] = 0.f;
        #pragma unroll
        for (int t = 0; t < 2; ++t)
            #pragma unroll
            for (int hf = 0; hf < 2; ++hf) {
                bf16x16 bk;                        // B = K^T: B[d][key] = K[key][d]
                #pragma unroll
                for (int v = 0; v < 8; ++v) {
                    int kl = (v < 4) ? (2 * v + 8 * g) : (16 + 2 * (v - 4) + 8 * g);
                    int d  = 32 * hf + kl;
                    bk[2 * v]     = ubf(Ksh[(16 * t + n16) * 64 + d]);
                    bk[2 * v + 1] = ubf(Ksh[(16 * t + n16) * 64 + d + 1]);
                }
                s[t] = __builtin_amdgcn_wmma_f32_16x16x32_bf16(
                    false, aq[hf], false, bk, (short)0, s[t], false, false);
            }

        // ---- rel gather + mask + scale (1/sqrt(64)) ----
        float p[2][8];
        #pragma unroll
        for (int t = 0; t < 2; ++t)
            #pragma unroll
            for (int r = 0; r < 8; ++r) {
                int row = r + 8 * g;
                int lq  = 16 * wave + row;
                int lk  = 16 * t + n16;
                int id  = idssh[lq * 32 + lk];
                float rel = relsh[lq * 32 + id];
                float m   = (float)msksh[lq * 32 + lk];
                p[t][r] = (s[t][r] + rel) * 0.125f - 10000.0f * (1.0f - m);
            }

        // ---- online softmax (rows live in 16-lane halves; xor<16 stays) ----
        #pragma unroll
        for (int r = 0; r < 8; ++r) {
            float rmax = fmaxf(p[0][r], p[1][r]);
            rmax = fmaxf(rmax, __shfl_xor(rmax, 1, 32));
            rmax = fmaxf(rmax, __shfl_xor(rmax, 2, 32));
            rmax = fmaxf(rmax, __shfl_xor(rmax, 4, 32));
            rmax = fmaxf(rmax, __shfl_xor(rmax, 8, 32));
            float mnew = fmaxf(mrun[r], rmax);
            float corr = __expf(mrun[r] - mnew);
            float e0 = __expf(p[0][r] - mnew);
            float e1 = __expf(p[1][r] - mnew);
            float rs = e0 + e1;
            rs += __shfl_xor(rs, 1, 32);
            rs += __shfl_xor(rs, 2, 32);
            rs += __shfl_xor(rs, 4, 32);
            rs += __shfl_xor(rs, 8, 32);
            lrun[r] = lrun[r] * corr + rs;
            mrun[r] = mnew;
            #pragma unroll
            for (int t = 0; t < 4; ++t) o[t][r] *= corr;
            p[0][r] = e0; p[1][r] = e1;
        }

        // ---- P: C-layout -> A-layout via wave-private LDS patch ----
        #pragma unroll
        for (int t = 0; t < 2; ++t)
            #pragma unroll
            for (int r = 0; r < 8; ++r)
                Psh[wave][(r + 8 * g) * 32 + 16 * t + n16] = f2bfu(p[t][r]);
        asm volatile("s_wait_dscnt 0" ::: "memory");   // LDS in-order per wave

        bf16x16 ap;
        #pragma unroll
        for (int v = 0; v < 8; ++v) {
            int kl = (v < 4) ? (2 * v + 8 * g) : (16 + 2 * (v - 4) + 8 * g);
            ap[2 * v]     = ubf(Psh[wave][n16 * 32 + kl]);
            ap[2 * v + 1] = ubf(Psh[wave][n16 * 32 + kl + 1]);
        }

        // ---- O += P(16x32) @ V(32x64): 4 d-tiles, full K=32 ----
        #pragma unroll
        for (int t = 0; t < 4; ++t) {
            bf16x16 bv;
            #pragma unroll
            for (int v = 0; v < 8; ++v) {
                int kl = (v < 4) ? (2 * v + 8 * g) : (16 + 2 * (v - 4) + 8 * g);
                bv[2 * v]     = ubf(Vsh[kl * 64 + 16 * t + n16]);
                bv[2 * v + 1] = ubf(Vsh[(kl + 1) * 64 + 16 * t + n16]);
            }
            o[t] = __builtin_amdgcn_wmma_f32_16x16x32_bf16(
                false, ap, false, bv, (short)0, o[t], false, false);
        }
    }

    // normalize and write ctx (bf16, [s][h*64+d])
    #pragma unroll
    for (int t = 0; t < 4; ++t)
        #pragma unroll
        for (int r = 0; r < 8; ++r) {
            int row = q0 + 16 * wave + r + 8 * g;
            int d   = 16 * t + n16;
            ctx[(size_t)row * 1024 + h * 64 + d] = f2bfu(o[t][r] / lrun[r]);
        }
}

// ---------------------------------------------------------------------------
// Host launcher
// ---------------------------------------------------------------------------
extern "C" void kernel_launch(void* const* d_in, const int* in_sizes, int n_in,
                              void* d_out, int out_size, void* d_ws, size_t ws_size,
                              hipStream_t stream) {
    (void)in_sizes; (void)n_in; (void)out_size; (void)ws_size;

    const float* x        = (const float*)d_in[0];
    const int*   att_mask = (const int*)  d_in[1];
    const int*   rel_ids  = (const int*)  d_in[2];
    const float* Wq = (const float*)d_in[3];
    const float* bq = (const float*)d_in[4];
    const float* Wk = (const float*)d_in[5];
    const float* bk = (const float*)d_in[6];
    const float* Wv = (const float*)d_in[7];
    const float* bv = (const float*)d_in[8];
    const float* rel_emb  = (const float*)d_in[9];
    const float* rel_bias = (const float*)d_in[10];
    const float* Wo = (const float*)d_in[11];
    const float* bo = (const float*)d_in[12];
    float* out = (float*)d_out;

    // workspace layout: q,k,v,ctx bf16 (4MB each) + all_rel f32 (4MB) = 20MB
    char* w = (char*)d_ws;
    const size_t MB4 = (size_t)4 * 1024 * 1024;
    unsigned short* qb  = (unsigned short*)(w);
    unsigned short* kb  = (unsigned short*)(w + 1 * MB4);
    unsigned short* vb  = (unsigned short*)(w + 2 * MB4);
    unsigned short* cb  = (unsigned short*)(w + 3 * MB4);
    float*          arl = (float*)         (w + 4 * MB4);

    dim3 blk(128);
    dim3 gGemm(1024 / 64, 2048 / 64);            // (N/64, M/64)

    gemm64x64<float, unsigned short><<<gGemm, blk, 0, stream>>>(x, Wq, bq, qb, 2048, 1024, 1024);
    gemm64x64<float, unsigned short><<<gGemm, blk, 0, stream>>>(x, Wk, bk, kb, 2048, 1024, 1024);
    gemm64x64<float, unsigned short><<<gGemm, blk, 0, stream>>>(x, Wv, bv, vb, 2048, 1024, 1024);

    rel_kernel<<<(2048 * 16 * 32) / 256, 256, 0, stream>>>(qb, rel_emb, rel_bias, arl);

    attn_kernel<<<dim3(2048 / 64, 16), blk, 0, stream>>>(qb, kb, vb, arl, rel_ids, att_mask, cb);

    gemm64x64<unsigned short, float><<<gGemm, blk, 0, stream>>>(cb, Wo, bo, out, 2048, 1024, 1024);
}
